// BasicBlock_37228776522124
// MI455X (gfx1250) — 8-bit matrix operand order repaired and checked by reading, compiled, not run on hardware
//
#include <hip/hip_runtime.h>
#include <cstdint>


typedef int   v16i __attribute__((ext_vector_type(16)));
typedef float v8f  __attribute__((ext_vector_type(8)));

#define NB      32
#define CCH     256
#define HW      56
#define HWHW    3136
#define PW      58
#define PP      3364
#define MROW    100352
#define NGROUPS 392
#define EPS     1e-5f

__device__ __forceinline__ uint8_t fp8sign(float v) {
    return v > 0.0f ? 0x38 : (v < 0.0f ? 0xB8 : 0x00);
}

__device__ __forceinline__ int permA(int c) {
    int k = c & 63;
    return (c & ~63) + ((k >> 3) & 1) * 32 + ((k >> 4) & 3) * 8 + (k & 7);
}

__global__ __launch_bounds__(256) void zero_pad_kernel(int4* __restrict__ p, size_t n16) {
    const size_t stride = (size_t)gridDim.x * blockDim.x;
    for (size_t i = (size_t)blockIdx.x * blockDim.x + threadIdx.x; i < n16; i += stride)
        p[i] = make_int4(0, 0, 0, 0);
}

__global__ __launch_bounds__(256) void pack_w_kernel(const float* __restrict__ w,
                                                     uint8_t* __restrict__ bpack,
                                                     float* __restrict__ alpha) {
    const int o = blockIdx.x;
    const int c = threadIdx.x;
    __shared__ float red[256];
    float s = 0.0f;
#pragma unroll
    for (int tap = 0; tap < 9; ++tap) {
        float v = w[((size_t)o * 256 + c) * 9 + tap];
        s += fabsf(v);
        bpack[((size_t)(tap * 256 + o)) * 256 + c] = fp8sign(v);
    }
    red[c] = s;
    __syncthreads();
    for (int st = 128; st > 0; st >>= 1) {
        if (c < st) red[c] += red[c + st];
        __syncthreads();
    }
    if (c == 0) alpha[o] = red[0] / 2304.0f;
}

__global__ __launch_bounds__(256) void pack_x_kernel(const float* __restrict__ x,
                                                     uint8_t* __restrict__ pack) {
    __shared__ float lds[32][33];
    const int tx = threadIdx.x, ty = threadIdx.y;
    const int wt = blockIdx.x & 1, ct = blockIdx.x >> 1;
    const int n  = blockIdx.y / HW, h = blockIdx.y % HW;
    const int w0 = wt * 32, c0 = ct * 32;
#pragma unroll
    for (int i = 0; i < 4; ++i) {
        int c = c0 + ty + 8 * i, w = w0 + tx;
        if (w < HW) lds[ty + 8 * i][tx] = x[((size_t)(n * CCH + c)) * HWHW + h * HW + w];
    }
    __syncthreads();
#pragma unroll
    for (int i = 0; i < 4; ++i) {
        int w = w0 + ty + 8 * i, c = c0 + tx;
        if (w < HW) {
            size_t base = ((size_t)(n * PP + (h + 1) * PW + (w + 1))) * CCH;
            pack[base + permA(c)] = fp8sign(lds[tx][ty + 8 * i]);
        }
    }
}

__global__ __launch_bounds__(256) void conv_bin_kernel(const uint8_t* __restrict__ apack,
                                                       const uint8_t* __restrict__ bpack,
                                                       const float* __restrict__ alpha,
                                                       float* __restrict__ y,
                                                       float* __restrict__ psum,
                                                       float* __restrict__ psq) {
    extern __shared__ char smem[];
    const int tid  = threadIdx.x;
    const int wave = tid >> 5;
    const int lane = tid & 31;
    const int l16  = lane & 15;
    const int half = lane >> 4;
    const int nstrip = blockIdx.x & 3;
    const int mgroup = blockIdx.x >> 2;
    const int nbase  = nstrip * 64;

    for (int i = tid; i < (9 * 64 * 256) / 16; i += 256) {
        int row = i >> 4, off = (i & 15) * 16;
        int tap = row >> 6, ocl = row & 63;
        *(int4*)(smem + (size_t)row * 256 + off) =
            *(const int4*)(bpack + ((size_t)(tap * 256 + nbase + ocl)) * 256 + off);
    }
    __syncthreads();

    const uint8_t* pbase[2];
#pragma unroll
    for (int t = 0; t < 2; ++t) {
        int m   = (mgroup * 16 + wave * 2 + t) * 16 + l16;
        int n   = m / HWHW;
        int rem = m % HWHW;
        int h   = rem / HW, w = rem % HW;
        pbase[t] = apack + ((size_t)(n * PP + (h + 1) * PW + (w + 1))) * CCH + half * 32;
    }

    v8f acc[2][4];
#pragma unroll
    for (int t = 0; t < 2; ++t)
#pragma unroll
        for (int j = 0; j < 4; ++j) acc[t][j] = (v8f)0.0f;

#pragma unroll
    for (int tap = 0; tap < 9; ++tap) {
        const int tapoff = ((tap / 3 - 1) * PW + (tap % 3 - 1)) * CCH;
#pragma unroll
        for (int cc = 0; cc < 2; ++cc) {
            v16i a[2];
#pragma unroll
            for (int t = 0; t < 2; ++t) {
                const char* ap = (const char*)pbase[t] + tapoff + cc * 128;
                int4 p0 = *(const int4*)(ap);
                int4 p1 = *(const int4*)(ap + 16);
                int4 p2 = *(const int4*)(ap + 64);
                int4 p3 = *(const int4*)(ap + 80);
                v16i av;
                av[0]  = p0.x; av[1]  = p0.y; av[2]  = p0.z; av[3]  = p0.w;
                av[4]  = p1.x; av[5]  = p1.y; av[6]  = p1.z; av[7]  = p1.w;
                av[8]  = p2.x; av[9]  = p2.y; av[10] = p2.z; av[11] = p2.w;
                av[12] = p3.x; av[13] = p3.y; av[14] = p3.z; av[15] = p3.w;
                a[t] = av;
            }
            v16i b[4];
#pragma unroll
            for (int j = 0; j < 4; ++j) {
                const char* bl = smem + ((size_t)(tap * 64 + j * 16 + l16)) * 256
                               + cc * 128 + half * 8;
                v16i bv;
#pragma unroll
                for (int g = 0; g < 8; ++g) {
                    int2 q = *(const int2*)(bl + g * 16);
                    bv[2 * g] = q.x; bv[2 * g + 1] = q.y;
                }
                b[j] = bv;
            }
#pragma unroll
            for (int j = 0; j < 4; ++j)
#pragma unroll
                for (int t = 0; t < 2; ++t)
                    acc[t][j] = __builtin_amdgcn_wmma_f32_16x16x128_fp8_fp8(
                        a[t], b[j], (short)0, acc[t][j], false, false);
        }
    }

    __syncthreads();
    float* pred = (float*)smem;
    const int cid = wave * 2 + half;
#pragma unroll
    for (int j = 0; j < 4; ++j) {
        const int   ocl = j * 16 + l16;
        const int   oc  = nbase + ocl;
        const float al  = alpha[oc];
        float s = 0.0f, ss = 0.0f;
#pragma unroll
        for (int t = 0; t < 2; ++t) {
            const int mb = (mgroup * 16 + wave * 2 + t) * 16 + half * 8;
#pragma unroll
            for (int r = 0; r < 8; ++r) {
                float v = acc[t][j][r] * al;
                y[(size_t)(mb + r) * CCH + oc] = v;
                s += v; ss += v * v;
            }
        }
        pred[cid * 64 + ocl]        = s;
        pred[1024 + cid * 64 + ocl] = ss;
    }
    __syncthreads();
    if (tid < 64) {
        float s = 0.0f, ss = 0.0f;
#pragma unroll
        for (int k = 0; k < 16; ++k) {
            s  += pred[k * 64 + tid];
            ss += pred[1024 + k * 64 + tid];
        }
        psum[(size_t)mgroup * 256 + nbase + tid] = s;
        psq [(size_t)mgroup * 256 + nbase + tid] = ss;
    }
}

__global__ __launch_bounds__(256) void bn_finalize_kernel(const float* __restrict__ psum,
                                                          const float* __restrict__ psq,
                                                          const float* __restrict__ gamma,
                                                          const float* __restrict__ beta,
                                                          float* __restrict__ scale,
                                                          float* __restrict__ bias) {
    const int c = threadIdx.x;
    float s = 0.0f, ss = 0.0f;
    for (int b = 0; b < NGROUPS; ++b) {
        s  += psum[(size_t)b * 256 + c];
        ss += psq [(size_t)b * 256 + c];
    }
    const float inv  = 1.0f / (float)MROW;
    const float mean = s * inv;
    const float var  = ss * inv - mean * mean;
    const float sc   = gamma[c] * rsqrtf(var + EPS);
    scale[c] = sc;
    bias[c]  = beta[c] - mean * sc;
}

__global__ __launch_bounds__(256) void bn_sign_pack_kernel(const float* __restrict__ y,
                                                           const float* __restrict__ scale,
                                                           const float* __restrict__ bias,
                                                           uint8_t* __restrict__ pack) {
    const int c  = threadIdx.x;
    const int po = permA(c);
    const float sc = scale[c], bi = bias[c];
    for (int m = blockIdx.x; m < MROW; m += gridDim.x) {
        int n = m / HWHW, rem = m % HWHW;
        int h = rem / HW,  w  = rem % HW;
        size_t base = ((size_t)(n * PP + (h + 1) * PW + (w + 1))) * CCH;
        float v = y[(size_t)m * CCH + c] * sc + bi;
        pack[base + po] = fp8sign(v);
    }
}

__global__ __launch_bounds__(256) void final_out_kernel(const float* __restrict__ y,
                                                        const float* __restrict__ scale,
                                                        const float* __restrict__ bias,
                                                        const float* __restrict__ x,
                                                        float* __restrict__ out) {
    __shared__ float lds[32][33];
    const int tx = threadIdx.x, ty = threadIdx.y;
    const int wt = blockIdx.x & 1, ct = blockIdx.x >> 1;
    const int n  = blockIdx.y / HW, h = blockIdx.y % HW;
    const int w0 = wt * 32, c0 = ct * 32;
#pragma unroll
    for (int i = 0; i < 4; ++i) {
        int w = w0 + ty + 8 * i, c = c0 + tx;
        if (w < HW) lds[ty + 8 * i][tx] = y[((size_t)(n * HWHW + h * HW + w)) * CCH + c];
    }
    __syncthreads();
#pragma unroll
    for (int i = 0; i < 4; ++i) {
        int c = c0 + ty + 8 * i, w = w0 + tx;
        if (w < HW) {
            size_t idx = ((size_t)(n * CCH + c)) * HWHW + h * HW + w;
            out[idx] = lds[tx][ty + 8 * i] * scale[c] + bias[c] + x[idx];
        }
    }
}

extern "C" void kernel_launch(void* const* d_in, const int* in_sizes, int n_in,
                              void* d_out, int out_size, void* d_ws, size_t ws_size,
                              hipStream_t stream) {
    const float* x      = (const float*)d_in[0];
    const float* w1     = (const float*)d_in[1];
    const float* gamma1 = (const float*)d_in[2];
    const float* beta1  = (const float*)d_in[3];
    const float* w2     = (const float*)d_in[4];
    const float* gamma2 = (const float*)d_in[5];
    const float* beta2  = (const float*)d_in[6];
    float* out = (float*)d_out;

    char* ws = (char*)d_ws;
    float*   y      = (float*)ws;   ws += (size_t)MROW * CCH * 4;
    uint8_t* pack   = (uint8_t*)ws; ws += (size_t)NB * PP * CCH;
    uint8_t* bpack1 = (uint8_t*)ws; ws += (size_t)9 * CCH * CCH;
    uint8_t* bpack2 = (uint8_t*)ws; ws += (size_t)9 * CCH * CCH;
    float*   psum   = (float*)ws;   ws += (size_t)NGROUPS * 256 * 4;
    float*   psq    = (float*)ws;   ws += (size_t)NGROUPS * 256 * 4;
    float*   alpha1 = (float*)ws;   ws += 1024;
    float*   alpha2 = (float*)ws;   ws += 1024;
    float*   scaleb = (float*)ws;   ws += 2048;
    float* bscale = scaleb, *bbias = scaleb + 256;
    (void)ws_size; (void)in_sizes; (void)n_in; (void)out_size;

    const dim3 tgrid(16, NB * HW), tblk(32, 8);
    const size_t conv_lds = 9 * 64 * 256;
    const size_t pad16 = ((size_t)NB * PP * CCH) / 16;

    zero_pad_kernel<<<2048, 256, 0, stream>>>((int4*)pack, pad16);
    pack_w_kernel<<<256, 256, 0, stream>>>(w1, bpack1, alpha1);
    pack_w_kernel<<<256, 256, 0, stream>>>(w2, bpack2, alpha2);
    pack_x_kernel<<<tgrid, tblk, 0, stream>>>(x, pack);

    conv_bin_kernel<<<1568, 256, conv_lds, stream>>>(pack, bpack1, alpha1, y, psum, psq);
    bn_finalize_kernel<<<1, 256, 0, stream>>>(psum, psq, gamma1, beta1, bscale, bbias);
    bn_sign_pack_kernel<<<4096, 256, 0, stream>>>(y, bscale, bbias, pack);

    conv_bin_kernel<<<1568, 256, conv_lds, stream>>>(pack, bpack2, alpha2, y, psum, psq);
    bn_finalize_kernel<<<1, 256, 0, stream>>>(psum, psq, gamma2, beta2, bscale, bbias);
    final_out_kernel<<<tgrid, tblk, 0, stream>>>(y, bscale, bbias, x, out);
}
